// FlowDecoder_89060441850076
// MI455X (gfx1250) — matrix operand order repaired and checked by reading, not compiled, not run on hardware
//
#include <hip/hip_runtime.h>
#include <hip/hip_bf16.h>
#include <stdint.h>

#define B_   2
#define S_   1024
#define D_   768
#define DC_  128
#define DT_  196
#define H_   12
#define L_   12
#define FF_  3072
#define DH_  64
#define MTOT (B_ * S_)   // 2048
#define EPS_ 1e-5f

typedef unsigned int u32;
typedef unsigned long long u64;
typedef __attribute__((ext_vector_type(16))) __bf16 v16bf;
typedef __attribute__((ext_vector_type(8)))  float  v8f;
typedef __attribute__((ext_vector_type(4)))  u32    v4u;
typedef __attribute__((ext_vector_type(8)))  int    v8i_t;
typedef __attribute__((ext_vector_type(4)))  int    v4i_t;

// ---------------------------------------------------------------------------
// Tensor Data Mover: 2D strided tile (bf16) global -> LDS, with row padding.
// D# layout per CDNA5 ISA ch.8. One wave issues; completion via TENSORcnt.
// ---------------------------------------------------------------------------
__device__ __forceinline__ u32 lds_off32(const void* p) {
  // generic LDS pointer: addr[31:0] is the LDS byte offset (aperture in MSBs)
  return (u32)(uintptr_t)p;
}

__device__ __forceinline__ void tdm_load_2d(u32 lds_off, const void* gp,
                                            u32 w_elems, u32 h_rows,
                                            u32 stride_elems, u32 pad_int_enc,
                                            u32 pad_amt_enc) {
  const u64 ga = (u64)(uintptr_t)gp;
  v4u g0;
  g0[0] = 1u;                                   // count=1, no gather
  g0[1] = lds_off;                              // lds_addr (bytes)
  g0[2] = (u32)ga;                              // global_addr[31:0]
  g0[3] = (u32)((ga >> 32) & 0x01ffffffu) | 0x80000000u;  // addr[56:32], type=2
  const u32 w0 = (1u << 16)                     // data_size = 1 -> 2 bytes
               | (1u << 20)                     // pad_enable
               | (pad_int_enc << 22) | (pad_amt_enc << 25);
  const u32 td0 = 0x7fffffffu, td1 = 0x7fffffffu;  // huge tensor dims (no clip)
  v8i_t g1;
  g1[0] = (int)w0;
  g1[1] = (int)((td0 & 0xffffu) << 16);                    // tensor_dim0 lo
  g1[2] = (int)((td0 >> 16) | ((td1 & 0xffffu) << 16));    // dim0 hi | dim1 lo
  g1[3] = (int)((td1 >> 16) | (w_elems << 16));            // dim1 hi | tile_dim0
  g1[4] = (int)(h_rows & 0xffffu);                         // tile_dim1 (dim2=0)
  g1[5] = (int)stride_elems;                               // dim0_stride lo32
  g1[6] = 0;                                               // stride hi, dim1_stride
  g1[7] = 0;
  const v4i_t z4 = {0, 0, 0, 0};
#if __clang_major__ >= 23
  const v8i_t z8 = {0, 0, 0, 0, 0, 0, 0, 0};
  __builtin_amdgcn_tensor_load_to_lds(g0, g1, z4, z4, z8, 0);
#else
  __builtin_amdgcn_tensor_load_to_lds(g0, g1, z4, z4, 0);
#endif
}

// ---------------------------------------------------------------------------
// WMMA fragment loaders (wave32, 16x16x32 bf16) -- contiguous k order: element i of lane half h holds k = 16h + i (A and B)
// ---------------------------------------------------------------------------
__device__ __forceinline__ v16bf load_a_frag(const __bf16* base, int ld) {
  const int lane = threadIdx.x & 31;
  const int m    = lane & 15;
  const int kb   = (lane >> 4) << 4;
  const __bf16* p = base + m * ld + kb;
  v16bf a;
#pragma unroll
  for (int i = 0; i < 8; ++i)  a[i] = p[i];
#pragma unroll
  for (int i = 8; i < 16; ++i) a[i] = p[i];
  return a;
}

// B tile stored [n][k] (e.g. W row-major (N,K) slabs)
__device__ __forceinline__ v16bf load_b_nk(const __bf16* base, int ld) {
  const int lane = threadIdx.x & 31;
  const int n    = lane & 15;
  const int koff = (lane >> 4) << 4;
  const __bf16* p = base + n * ld + koff;
  v16bf b;
#pragma unroll
  for (int i = 0; i < 16; ++i) b[i] = p[i];
  return b;
}

// B tile stored [k][n] (e.g. V tile [key][dim])
__device__ __forceinline__ v16bf load_b_kn(const __bf16* base, int ld) {
  const int lane = threadIdx.x & 31;
  const int n    = lane & 15;
  const int koff = (lane >> 4) << 4;
  v16bf b;
#pragma unroll
  for (int i = 0; i < 16; ++i) b[i] = base[(koff + i) * ld + n];
  return b;
}

#define WMMA_BF16(A, Bf, C)                                               \
  __builtin_amdgcn_wmma_f32_16x16x32_bf16(false, (A), false, (Bf),        \
                                          (short)0, (C), false, false)

// ---------------------------------------------------------------------------
// GEMM: C = [gelu](A(MxK,bf16) * W(NxK,bf16)^T + bias) [+ res]; out f32/bf16
// 128x128 block tile, 8 waves (each 32x64), K-step 32, TDM double buffering.
// M,N multiples of 128; K multiple of 32 at all call sites.
// ---------------------------------------------------------------------------
template <bool BIAS, bool RES, bool GELU, bool OUTBF>
__global__ __launch_bounds__(256) void gemm_tdm(
    const __bf16* __restrict__ A, const __bf16* __restrict__ W,
    const float* __restrict__ bias, const float* __restrict__ res,
    void* __restrict__ Cv, int M, int N, int K) {
  __shared__ __bf16 As[2][128 * 40];
  __shared__ __bf16 Ws[2][128 * 40];
  const int n0   = blockIdx.x * 128;
  const int m0   = blockIdx.y * 128;
  const int wv   = threadIdx.x >> 5;
  const int lane = threadIdx.x & 31;
  const int mrow = (wv & 3) * 32;
  const int nc0  = (wv >> 2) * 64;

  v8f acc[2][4] = {};
  const int nk = K / 32;

  if (wv == 0) {
    tdm_load_2d(lds_off32(&As[0][0]), A + (size_t)m0 * K, 32, 128, K, 3, 3);
    tdm_load_2d(lds_off32(&Ws[0][0]), W + (size_t)n0 * K, 32, 128, K, 3, 3);
  }
  for (int i = 0; i < nk; ++i) {
    const int cur = i & 1;
    if (wv == 0) {
      if (i + 1 < nk) {
        tdm_load_2d(lds_off32(&As[cur ^ 1][0]),
                    A + (size_t)m0 * K + (i + 1) * 32, 32, 128, K, 3, 3);
        tdm_load_2d(lds_off32(&Ws[cur ^ 1][0]),
                    W + (size_t)n0 * K + (i + 1) * 32, 32, 128, K, 3, 3);
        __builtin_amdgcn_s_wait_tensorcnt(2);
      } else {
        __builtin_amdgcn_s_wait_tensorcnt(0);
      }
    }
    __syncthreads();
    const __bf16* ab = &As[cur][0];
    const __bf16* wb = &Ws[cur][0];
    const v16bf af0 = load_a_frag(ab + mrow * 40, 40);
    const v16bf af1 = load_a_frag(ab + (mrow + 16) * 40, 40);
#pragma unroll
    for (int nt = 0; nt < 4; ++nt) {
      const v16bf bfrag = load_b_nk(wb + (nc0 + nt * 16) * 40, 40);
      acc[0][nt] = WMMA_BF16(af0, bfrag, acc[0][nt]);
      acc[1][nt] = WMMA_BF16(af1, bfrag, acc[1][nt]);
    }
    __syncthreads();
  }

  const int ncol = lane & 15;
  const int half = lane >> 4;
#pragma unroll
  for (int s = 0; s < 2; ++s) {
#pragma unroll
    for (int nt = 0; nt < 4; ++nt) {
#pragma unroll
      for (int r = 0; r < 8; ++r) {
        const int gm = m0 + mrow + s * 16 + r + 8 * half;
        const int gn = n0 + nc0 + nt * 16 + ncol;
        float v = acc[s][nt][r];
        if (BIAS) v += bias[gn];
        if (GELU) v = 0.5f * v * (1.0f + erff(v * 0.70710678118654752f));
        if (RES)  v += res[(size_t)gm * N + gn];
        if (OUTBF) ((__bf16*)Cv)[(size_t)gm * N + gn] = (__bf16)v;
        else       ((float*)Cv)[(size_t)gm * N + gn]  = v;
      }
    }
  }
}

// ---------------------------------------------------------------------------
// Flash attention on bf16 qkv: block = (64 q-rows, head, batch), 4 waves.
// Q loaded once via TDM; K/V double-buffered via TDM. Online softmax.
// ---------------------------------------------------------------------------
__global__ __launch_bounds__(128) void flash_attn(
    const __bf16* __restrict__ qkv, __bf16* __restrict__ out) {
  const int qt   = blockIdx.x;
  const int h    = blockIdx.y;
  const int b    = blockIdx.z;
  const int wv   = threadIdx.x >> 5;
  const int lane = threadIdx.x & 31;
  const int ncol = lane & 15;
  const int half = lane >> 4;

  __shared__ __bf16 Qs[64 * 72];
  __shared__ __bf16 Ks[2][64 * 72];
  __shared__ __bf16 Vs[2][64 * 72];
  __shared__ __bf16 Ps[4][16 * 72];

  const int    row0  = qt * 64;
  const size_t basep = (size_t)b * S_ * (3 * D_);
  const __bf16* qp = qkv + basep + (size_t)row0 * (3 * D_) + h * DH_;
  const __bf16* kp = qkv + basep + D_ + h * DH_;
  const __bf16* vp = qkv + basep + 2 * D_ + h * DH_;

  if (wv == 0) {
    tdm_load_2d(lds_off32(&Qs[0]),    qp, 64, 64, 3 * D_, 4, 3);
    tdm_load_2d(lds_off32(&Ks[0][0]), kp, 64, 64, 3 * D_, 4, 3);
    tdm_load_2d(lds_off32(&Vs[0][0]), vp, 64, 64, 3 * D_, 4, 3);
    __builtin_amdgcn_s_wait_tensorcnt(0);
  }
  __syncthreads();

  const v16bf qa0 = load_a_frag(Qs + (wv * 16) * 72, 72);
  const v16bf qa1 = load_a_frag(Qs + (wv * 16) * 72 + 32, 72);

  v8f   o_acc[4] = {};
  float run_m[8], run_l[8];
#pragma unroll
  for (int r = 0; r < 8; ++r) { run_m[r] = -1e30f; run_l[r] = 0.0f; }
  const float scale = 0.125f;  // DH^-0.5

  for (int j = 0; j < S_ / 64; ++j) {
    const int cur = j & 1;
    if (wv == 0) {
      if (j + 1 < S_ / 64) {
        tdm_load_2d(lds_off32(&Ks[cur ^ 1][0]),
                    kp + (size_t)(j + 1) * 64 * (3 * D_), 64, 64, 3 * D_, 4, 3);
        tdm_load_2d(lds_off32(&Vs[cur ^ 1][0]),
                    vp + (size_t)(j + 1) * 64 * (3 * D_), 64, 64, 3 * D_, 4, 3);
        __builtin_amdgcn_s_wait_tensorcnt(2);
      } else {
        __builtin_amdgcn_s_wait_tensorcnt(0);
      }
    }
    __syncthreads();
    const __bf16* ks = &Ks[cur][0];
    const __bf16* vs = &Vs[cur][0];

    v8f s_acc[4] = {};
#pragma unroll
    for (int kt = 0; kt < 4; ++kt) {
      const v16bf kb0 = load_b_nk(ks + (kt * 16) * 72, 72);
      const v16bf kb1 = load_b_nk(ks + (kt * 16) * 72 + 32, 72);
      s_acc[kt] = WMMA_BF16(qa0, kb0, s_acc[kt]);
      s_acc[kt] = WMMA_BF16(qa1, kb1, s_acc[kt]);
    }

    float nm[8], corr[8], rsum[8];
#pragma unroll
    for (int r = 0; r < 8; ++r) {
      float mx = -1e30f;
#pragma unroll
      for (int kt = 0; kt < 4; ++kt) {
        s_acc[kt][r] *= scale;
        mx = fmaxf(mx, s_acc[kt][r]);
      }
#pragma unroll
      for (int msk = 1; msk < 16; msk <<= 1)
        mx = fmaxf(mx, __shfl_xor(mx, msk, 32));
      nm[r]    = fmaxf(run_m[r], mx);
      corr[r]  = __expf(run_m[r] - nm[r]);
      run_m[r] = nm[r];
      rsum[r]  = 0.0f;
    }
#pragma unroll
    for (int kt = 0; kt < 4; ++kt) {
#pragma unroll
      for (int r = 0; r < 8; ++r) {
        const float p = __expf(s_acc[kt][r] - nm[r]);
        rsum[r] += p;
        Ps[wv][(r + 8 * half) * 72 + kt * 16 + ncol] = (__bf16)p;
      }
    }
#pragma unroll
    for (int r = 0; r < 8; ++r) {
#pragma unroll
      for (int msk = 1; msk < 16; msk <<= 1)
        rsum[r] += __shfl_xor(rsum[r], msk, 32);
      run_l[r] = run_l[r] * corr[r] + rsum[r];
#pragma unroll
      for (int ct = 0; ct < 4; ++ct) o_acc[ct][r] *= corr[r];
    }

    const v16bf pa0 = load_a_frag(&Ps[wv][0], 72);
    const v16bf pa1 = load_a_frag(&Ps[wv][0] + 32, 72);
#pragma unroll
    for (int ct = 0; ct < 4; ++ct) {
      const v16bf vb0 = load_b_kn(vs + ct * 16, 72);
      const v16bf vb1 = load_b_kn(vs + 32 * 72 + ct * 16, 72);
      o_acc[ct] = WMMA_BF16(pa0, vb0, o_acc[ct]);
      o_acc[ct] = WMMA_BF16(pa1, vb1, o_acc[ct]);
    }
    __syncthreads();
  }

#pragma unroll
  for (int ct = 0; ct < 4; ++ct) {
#pragma unroll
    for (int r = 0; r < 8; ++r) {
      const int grow = row0 + wv * 16 + r + 8 * half;
      out[((size_t)b * S_ + grow) * D_ + h * DH_ + ct * 16 + ncol] =
          (__bf16)(o_acc[ct][r] / run_l[r]);
    }
  }
}

// ---------------------------------------------------------------------------
// Elementwise / small kernels
// ---------------------------------------------------------------------------
__global__ void cvt_f32_bf16_kernel(const float* __restrict__ in,
                                    __bf16* __restrict__ out, int n) {
  const int stride = gridDim.x * blockDim.x;
  for (int i = blockIdx.x * blockDim.x + threadIdx.x; i < n; i += stride)
    out[i] = (__bf16)in[i];
}

__global__ void time_emb_kernel(const float* __restrict__ t,
                                const float* __restrict__ Wt,
                                const float* __restrict__ bt,
                                float* __restrict__ te) {
  const int b  = threadIdx.x / DC_;
  const int dc = threadIdx.x % DC_;
  const float tv = t[b];
  float s = 0.0f;
  for (int i = 0; i < DT_ / 2; ++i) {
    const float fr = __expf(-logf(10000.0f) * (float)i / (float)(DT_ / 2 - 1));
    const float e  = tv * fr;
    s += Wt[dc * DT_ + i] * __sinf(e) + Wt[dc * DT_ + DT_ / 2 + i] * __cosf(e);
  }
  te[b * DC_ + dc] = s + bt[dc];
}

__global__ void cond_silu_kernel(const float* __restrict__ ct,
                                 const float* __restrict__ te,
                                 const float* __restrict__ semb,
                                 const int* __restrict__ sidx,
                                 __bf16* __restrict__ cg) {
  const int idx = blockIdx.x * blockDim.x + threadIdx.x;
  const int dc  = idx % DC_;
  const int b   = (idx / DC_) / S_;
  const float v = ct[idx] + te[b * DC_ + dc] + semb[sidx[0] * DC_ + dc];
  cg[idx] = (__bf16)(v / (1.0f + __expf(-v)));
}

__global__ void input_proj_kernel(const float* __restrict__ xt,
                                  const float* __restrict__ w,
                                  const float* __restrict__ bias,
                                  float* __restrict__ x) {
  const int idx = blockIdx.x * blockDim.x + threadIdx.x;
  const int m = idx / D_, n = idx % D_;
  float v = bias[n];
#pragma unroll
  for (int c = 0; c < 3; ++c) v += xt[m * 3 + c] * w[n * 3 + c];
  x[idx] = v;
}

__device__ __forceinline__ void block_reduce(float* red, float& v) {
  red[threadIdx.x] = v;
  __syncthreads();
  for (int st = 128; st > 0; st >>= 1) {
    if (threadIdx.x < st) red[threadIdx.x] += red[threadIdx.x + st];
    __syncthreads();
  }
  v = red[0];
  __syncthreads();
}

// out(bf16) = ln(x) * (1 + g) + b, g/b per-row from gb (M, 2D)
__global__ __launch_bounds__(256) void ln_adaln_kernel(
    const float* __restrict__ x, const float* __restrict__ gb,
    __bf16* __restrict__ out) {
  __shared__ float red[256];
  const int row = blockIdx.x;
  const float* xr = x + (size_t)row * D_;
  float s = 0.0f;
  for (int i = threadIdx.x; i < D_; i += 256) s += xr[i];
  block_reduce(red, s);
  const float mu = s / (float)D_;
  float v = 0.0f;
  for (int i = threadIdx.x; i < D_; i += 256) {
    const float d = xr[i] - mu;
    v += d * d;
  }
  block_reduce(red, v);
  const float rstd = rsqrtf(v / (float)D_ + EPS_);
  const float* g = gb + (size_t)row * (2 * D_);
  const float* b = g + D_;
  for (int i = threadIdx.x; i < D_; i += 256)
    out[(size_t)row * D_ + i] =
        (__bf16)((xr[i] - mu) * rstd * (1.0f + g[i]) + b[i]);
}

__global__ __launch_bounds__(256) void ln_final_kernel(
    const float* __restrict__ x, const float* __restrict__ gamma,
    const float* __restrict__ beta, float* __restrict__ out) {
  __shared__ float red[256];
  const int row = blockIdx.x;
  const float* xr = x + (size_t)row * D_;
  float s = 0.0f;
  for (int i = threadIdx.x; i < D_; i += 256) s += xr[i];
  block_reduce(red, s);
  const float mu = s / (float)D_;
  float v = 0.0f;
  for (int i = threadIdx.x; i < D_; i += 256) {
    const float d = xr[i] - mu;
    v += d * d;
  }
  block_reduce(red, v);
  const float rstd = rsqrtf(v / (float)D_ + EPS_);
  for (int i = threadIdx.x; i < D_; i += 256)
    out[(size_t)row * D_ + i] = (xr[i] - mu) * rstd * gamma[i] + beta[i];
}

__global__ void out_proj_kernel(const float* __restrict__ h,
                                const float* __restrict__ w,
                                const float* __restrict__ bias,
                                float* __restrict__ out) {
  const int idx = blockIdx.x * blockDim.x + threadIdx.x;
  if (idx >= MTOT * 3) return;
  const int m = idx / 3, j = idx % 3;
  float v = bias[j];
  for (int n = 0; n < D_; ++n) v += h[(size_t)m * D_ + n] * w[j * D_ + n];
  out[idx] = v;
}

// ---------------------------------------------------------------------------
// Host launcher
// ---------------------------------------------------------------------------
extern "C" void kernel_launch(void* const* d_in, const int* in_sizes, int n_in,
                              void* d_out, int out_size, void* d_ws,
                              size_t ws_size, hipStream_t stream) {
  const float* x_t          = (const float*)d_in[0];
  const float* t            = (const float*)d_in[1];
  const float* z            = (const float*)d_in[2];
  const int*   scale_idx    = (const int*)d_in[3];
  const float* input_proj_w = (const float*)d_in[4];
  const float* input_proj_b = (const float*)d_in[5];
  const float* time_proj_w  = (const float*)d_in[6];
  const float* time_proj_b  = (const float*)d_in[7];
  const float* scale_emb    = (const float*)d_in[8];
  const float* cond_proj_w  = (const float*)d_in[9];
  const float* cond_proj_b  = (const float*)d_in[10];
  const float* adaln1_w     = (const float*)d_in[11];
  const float* adaln1_b     = (const float*)d_in[12];
  const float* qkv_w        = (const float*)d_in[13];
  const float* attn_out_w   = (const float*)d_in[14];
  const float* adaln2_w     = (const float*)d_in[15];
  const float* adaln2_b     = (const float*)d_in[16];
  const float* ffn1_w       = (const float*)d_in[17];
  const float* ffn1_b       = (const float*)d_in[18];
  const float* ffn2_w       = (const float*)d_in[19];
  const float* ffn2_b       = (const float*)d_in[20];
  const float* norm_out_g   = (const float*)d_in[21];
  const float* norm_out_b   = (const float*)d_in[22];
  const float* out_proj_w   = (const float*)d_in[23];
  const float* out_proj_b   = (const float*)d_in[24];
  float* outp = (float*)d_out;

  char* wsp = (char*)d_ws;
  auto carve = [&](size_t bytes) {
    char* r = wsp;
    wsp += (bytes + 255) & ~(size_t)255;
    return (void*)r;
  };
  float*  X     = (float*)carve((size_t)MTOT * D_ * 4);
  float*  GB    = (float*)carve((size_t)MTOT * 2 * D_ * 4);
  float*  CT    = (float*)carve((size_t)MTOT * DC_ * 4);
  float*  LNF   = (float*)carve((size_t)MTOT * D_ * 4);
  float*  TE    = (float*)carve((size_t)B_ * DC_ * 4);
  __bf16* Zbf   = (__bf16*)carve((size_t)MTOT * DC_ * 2);
  __bf16* CGbf  = (__bf16*)carve((size_t)MTOT * DC_ * 2);
  __bf16* Hbbf  = (__bf16*)carve((size_t)MTOT * D_ * 2);
  __bf16* QKVbf = (__bf16*)carve((size_t)MTOT * 3 * D_ * 2);
  __bf16* AObf  = (__bf16*)carve((size_t)MTOT * D_ * 2);
  __bf16* FFBbf = (__bf16*)carve((size_t)MTOT * FF_ * 2);
  __bf16* WBF   = (__bf16*)carve((size_t)FF_ * D_ * 2);

  const int MB = MTOT / 128;  // 16 row-blocks
  auto cvt = [&](const float* src, __bf16* dst, int n) {
    int blocks = (n + 255) / 256;
    if (blocks > 2048) blocks = 2048;
    cvt_f32_bf16_kernel<<<blocks, 256, 0, stream>>>(src, dst, n);
  };

  // conditioning
  time_emb_kernel<<<1, B_ * DC_, 0, stream>>>(t, time_proj_w, time_proj_b, TE);
  cvt(z, Zbf, MTOT * DC_);
  cvt(cond_proj_w, WBF, DC_ * DC_);
  gemm_tdm<true, false, false, false><<<dim3(1, MB), 256, 0, stream>>>(
      Zbf, WBF, cond_proj_b, nullptr, CT, MTOT, DC_, DC_);
  cond_silu_kernel<<<(MTOT * DC_) / 256, 256, 0, stream>>>(CT, TE, scale_emb,
                                                           scale_idx, CGbf);
  input_proj_kernel<<<(MTOT * D_) / 256, 256, 0, stream>>>(
      x_t, input_proj_w, input_proj_b, X);

  for (int l = 0; l < L_; ++l) {
    // AdaLN 1 + apply
    cvt(adaln1_w + (size_t)l * 2 * D_ * DC_, WBF, 2 * D_ * DC_);
    gemm_tdm<true, false, false, false>
        <<<dim3((2 * D_) / 128, MB), 256, 0, stream>>>(
            CGbf, WBF, adaln1_b + (size_t)l * 2 * D_, nullptr, GB, MTOT,
            2 * D_, DC_);
    ln_adaln_kernel<<<MTOT, 256, 0, stream>>>(X, GB, Hbbf);
    // QKV (bf16 out)
    cvt(qkv_w + (size_t)l * 3 * D_ * D_, WBF, 3 * D_ * D_);
    gemm_tdm<false, false, false, true>
        <<<dim3((3 * D_) / 128, MB), 256, 0, stream>>>(
            Hbbf, WBF, nullptr, nullptr, QKVbf, MTOT, 3 * D_, D_);
    // attention
    flash_attn<<<dim3(S_ / 64, H_, B_), 128, 0, stream>>>(QKVbf, AObf);
    // out proj + residual
    cvt(attn_out_w + (size_t)l * D_ * D_, WBF, D_ * D_);
    gemm_tdm<false, true, false, false>
        <<<dim3(D_ / 128, MB), 256, 0, stream>>>(AObf, WBF, nullptr, X, X,
                                                 MTOT, D_, D_);
    // AdaLN 2 + apply
    cvt(adaln2_w + (size_t)l * 2 * D_ * DC_, WBF, 2 * D_ * DC_);
    gemm_tdm<true, false, false, false>
        <<<dim3((2 * D_) / 128, MB), 256, 0, stream>>>(
            CGbf, WBF, adaln2_b + (size_t)l * 2 * D_, nullptr, GB, MTOT,
            2 * D_, DC_);
    ln_adaln_kernel<<<MTOT, 256, 0, stream>>>(X, GB, Hbbf);
    // FFN
    cvt(ffn1_w + (size_t)l * FF_ * D_, WBF, FF_ * D_);
    gemm_tdm<true, false, true, true>
        <<<dim3(FF_ / 128, MB), 256, 0, stream>>>(
            Hbbf, WBF, ffn1_b + (size_t)l * FF_, nullptr, FFBbf, MTOT, FF_,
            D_);
    cvt(ffn2_w + (size_t)l * D_ * FF_, WBF, D_ * FF_);
    gemm_tdm<true, true, false, false>
        <<<dim3(D_ / 128, MB), 256, 0, stream>>>(
            FFBbf, WBF, ffn2_b + (size_t)l * D_, X, X, MTOT, D_, FF_);
  }

  ln_final_kernel<<<MTOT, 256, 0, stream>>>(X, norm_out_g, norm_out_b, LNF);
  out_proj_kernel<<<(MTOT * 3 + 127) / 128, 128, 0, stream>>>(
      LNF, out_proj_w, out_proj_b, outp);
  (void)in_sizes; (void)n_in; (void)out_size; (void)ws_size;
}
